// Qwen3VLMoeTextSparseMoeBlock_37812892074072
// MI455X (gfx1250) — matrix operand order repaired and checked by reading, not compiled, not run on hardware
//
#include <hip/hip_runtime.h>
#include <hip/hip_bf16.h>

// ---------------- problem constants ----------------
#define T_TOK      2048      // 4*512 tokens
#define DHID       2048      // hidden
#define NEXP       32
#define TOPK       4
#define FINT       768       // intermediate
#define TWO_F      1536
#define MAXSLOTS   8704      // 8192 real + 32*16 padding headroom
#define MAXTILES   (MAXSLOTS/16)   // 544

typedef __attribute__((ext_vector_type(16))) __bf16 v16bf;
typedef __attribute__((ext_vector_type(8)))  __bf16 v8bf;
typedef __attribute__((ext_vector_type(8)))  float  v8f;

// fp32 -> bf16, round-to-nearest-even
__device__ __forceinline__ __bf16 f2bf(float f) {
  union { float f; unsigned u; } v; v.f = f;
  unsigned r = v.u + 0x7FFFu + ((v.u >> 16) & 1u);
  unsigned short h = (unsigned short)(r >> 16);
  return __builtin_bit_cast(__bf16, h);
}

// ---- WMMA bf16 fragment loaders (CDNA5 16x16x32 layouts, ISA 7.12.2) ----
// A (16x32, row in lane): lanes 0-15 row M=L hold K = k0+[0..7] and k0+[16..23];
// lanes 16-31 row M=L-16 hold K = k0+[8..15] and k0+[24..31].
__device__ __forceinline__ v16bf load_a_frag(const __bf16* row, int k0, int lane) {
  int base = k0 + ((lane >> 4) << 3);
  v8bf lo = *(const v8bf*)(row + base);
  v8bf hi = *(const v8bf*)(row + base + 16);
  v16bf r;
#pragma unroll
  for (int i = 0; i < 8; ++i) { r[i] = lo[i]; r[i + 8] = hi[i]; }
  return r;
}
// B (32x16, col in lane): lanes 0-15 col N hold K = k0+[0..7] and k0+[16..23];
// lanes 16-31 hold K = k0+[8..15] and k0+[24..31] (same order as A).
__device__ __forceinline__ v16bf load_b_frag(const __bf16* col, int k0, int lane) {
  int base = k0 + ((lane >> 4) << 3);
  v8bf lo = *(const v8bf*)(col + base);
  v8bf hi = *(const v8bf*)(col + base + 16);
  v16bf r;
#pragma unroll
  for (int i = 0; i < 8; ++i) { r[i] = lo[i]; r[i + 8] = hi[i]; }
  return r;
}

// ================= 1. router: one wave32 per token =================
__global__ __launch_bounds__(256, 1)
void router_kernel(const float* __restrict__ x, const float* __restrict__ gate,
                   int* __restrict__ topk_idx, float* __restrict__ topk_w) {
  int wave = threadIdx.x >> 5, lane = threadIdx.x & 31;
  int t = blockIdx.x * 8 + wave;
  if (t >= T_TOK) return;
  const float* xr = x + (size_t)t * DHID;
  float acc = 0.f;
  for (int d = 0; d < DHID; ++d)
    acc = fmaf(xr[d], gate[d * NEXP + lane], acc);   // lane == expert id
  // softmax across the 32 lanes
  float m = acc;
  for (int off = 16; off; off >>= 1) m = fmaxf(m, __shfl_xor(m, off, 32));
  float p = __expf(acc - m);
  float s = p;
  for (int off = 16; off; off >>= 1) s += __shfl_xor(s, off, 32);
  p /= s;
  // top-4 (ties -> lowest index, matching lax.top_k)
  float pv = p, wsum = 0.f, wk[TOPK]; int ik[TOPK];
#pragma unroll
  for (int k = 0; k < TOPK; ++k) {
    float bv = pv; int bi = lane;
    for (int off = 16; off; off >>= 1) {
      float ov = __shfl_xor(bv, off, 32); int oi = __shfl_xor(bi, off, 32);
      if (ov > bv || (ov == bv && oi < bi)) { bv = ov; bi = oi; }
    }
    wk[k] = bv; ik[k] = bi; wsum += bv;
    if (lane == bi) pv = -1.f;
  }
  if (lane < TOPK) {
    topk_idx[t * TOPK + lane] = ik[lane];
    topk_w[t * TOPK + lane]   = wk[lane] / wsum;
  }
}

// ================= 2a. per-expert token counts =================
__global__ void count_kernel(const int* __restrict__ topk_idx, int* __restrict__ counts) {
  int e = blockIdx.x, tid = threadIdx.x;
  __shared__ int c;
  if (tid == 0) c = 0;
  __syncthreads();
  int local = 0;
  for (int t = tid; t < T_TOK; t += 256)
#pragma unroll
    for (int k = 0; k < TOPK; ++k)
      if (topk_idx[t * TOPK + k] == e) ++local;
  atomicAdd(&c, local);
  __syncthreads();
  if (tid == 0) counts[e] = c;
}

// ================= 2b. padded exclusive scan + tile->expert map =================
__global__ void scan_kernel(const int* __restrict__ counts, int* __restrict__ base,
                            int* __restrict__ tile_expert) {
  int lane = threadIdx.x;             // 32 threads, one wave
  for (int i = lane; i < MAXTILES; i += 32) tile_expert[i] = -1;
  int padded = (counts[lane] + 15) & ~15;
  int xv = padded;
#pragma unroll
  for (int off = 1; off < 32; off <<= 1) {
    int y = __shfl_up(xv, off, 32);
    if (lane >= off) xv += y;
  }
  int excl = xv - padded;
  base[lane] = excl;
  if (lane == 31) base[32] = xv;      // padded total
  int t0 = excl >> 4, nt = padded >> 4;
  for (int i = 0; i < nt; ++i) tile_expert[t0 + i] = lane;
}

// ================= 2c. stable (token-ascending) slot fill =================
__global__ __launch_bounds__(256, 1)
void fill_kernel(const int* __restrict__ topk_idx, const float* __restrict__ topk_w,
                 const int* __restrict__ base, int* __restrict__ slot_tok,
                 int* __restrict__ slot_k, float* __restrict__ slot_w) {
  int e = blockIdx.x, tid = threadIdx.x;
  __shared__ int scan[256];
  __shared__ int running;
  if (tid == 0) running = 0;
  __syncthreads();
  int b = base[e];
  for (int t0 = 0; t0 < T_TOK; t0 += 256) {
    int t = t0 + tid, myk = -1;
#pragma unroll
    for (int k = 0; k < TOPK; ++k)
      if (topk_idx[t * TOPK + k] == e) myk = k;
    int flag = (myk >= 0);
    scan[tid] = flag;
    __syncthreads();
    for (int off = 1; off < 256; off <<= 1) {
      int v = (tid >= off) ? scan[tid - off] : 0;
      __syncthreads();
      scan[tid] += v;
      __syncthreads();
    }
    int incl = scan[tid], total = scan[255];
    if (flag) {
      int pos = b + running + incl - 1;
      slot_tok[pos] = t; slot_k[pos] = myk; slot_w[pos] = topk_w[t * TOPK + myk];
    }
    __syncthreads();
    if (tid == 0) running += total;
    __syncthreads();
  }
  int cnt = running;
  int padded = (cnt + 15) & ~15;
  if (tid < padded - cnt) {
    int pos = b + cnt + tid;
    slot_tok[pos] = 0; slot_k[pos] = -1; slot_w[pos] = 0.f;
  }
}

// ================= 3. fp32 -> bf16 conversions (weights pre-transposed) =======
__global__ void convert_x_kernel(const float* __restrict__ x, __bf16* __restrict__ xbf) {
  size_t o = (size_t)blockIdx.x * 256 + threadIdx.x;           // T*D elems
  xbf[o] = f2bf(x[o]);
}
// gate_up_proj (E, D, 2F) -> (E, 2F, D): block = (e*2F + f), contiguous-D rows out
__global__ void convert_gup_kernel(const float* __restrict__ w, __bf16* __restrict__ wt) {
  int ef = blockIdx.x;            // 0 .. E*2F-1
  int e = ef / TWO_F, f = ef % TWO_F;
  const float* src = w + (size_t)e * DHID * TWO_F + f;
  __bf16* dst = wt + (size_t)ef * DHID;
  for (int d = threadIdx.x; d < DHID; d += 256)
    dst[d] = f2bf(src[(size_t)d * TWO_F]);
}
// down_proj (E, F, D) -> (E, D, F)
__global__ void convert_down_kernel(const float* __restrict__ w, __bf16* __restrict__ wt) {
  int ed = blockIdx.x;            // 0 .. E*D-1
  int e = ed / DHID, d = ed % DHID;
  const float* src = w + (size_t)e * FINT * DHID + d;
  __bf16* dst = wt + (size_t)ed * FINT;
  for (int f = threadIdx.x; f < FINT; f += 256)
    dst[f] = f2bf(src[(size_t)f * DHID]);
}

// ================= 4. grouped GEMM1 + SiLU (bf16 WMMA, f32 acc) ===============
// grid.x = tile (16 slots), grid.y = f-chunk of 128 (6 chunks covering FINT)
__global__ __launch_bounds__(256, 1)
void moe_gateup_kernel(const __bf16* __restrict__ xbf, const __bf16* __restrict__ gupT,
                       const int* __restrict__ tile_expert, const int* __restrict__ slot_tok,
                       __bf16* __restrict__ gated) {
  __shared__ __bf16 As[16 * DHID];                 // 64 KB of the 320 KB WGP LDS
  int tile = blockIdx.x;
  int e = tile_expert[tile];
  if (e < 0) return;
  int tid = threadIdx.x, slot0 = tile * 16;
  for (int i = tid; i < 16 * 256; i += 256) {      // stage A tile (16 x 2048 bf16)
    int r = i >> 8, c8 = (i & 255) << 3;
    int tok = slot_tok[slot0 + r];
    *(v8bf*)(As + r * DHID + c8) = *(const v8bf*)(xbf + (size_t)tok * DHID + c8);
  }
  __syncthreads();
  int wave = tid >> 5, lane = tid & 31;
  int f = blockIdx.y * 128 + wave * 16 + (lane & 15);
  const __bf16* colg = gupT + ((size_t)e * TWO_F + f) * DHID;       // gate half
  const __bf16* colu = colg + (size_t)FINT * DHID;                  // up half
  const __bf16* arow = As + (size_t)(lane & 15) * DHID;
  v8f cg = {}; v8f cu = {};
  for (int k0 = 0; k0 < DHID; k0 += 32) {
    __builtin_prefetch(colg + k0 + 256, 0, 0);     // global_prefetch_b8
    __builtin_prefetch(colu + k0 + 256, 0, 0);
    v16bf a  = load_a_frag(arow, k0, lane);
    v16bf bg = load_b_frag(colg, k0, lane);
    v16bf bu = load_b_frag(colu, k0, lane);
    cg = __builtin_amdgcn_wmma_f32_16x16x32_bf16(false, a, false, bg, (short)0, cg, false, false);
    cu = __builtin_amdgcn_wmma_f32_16x16x32_bf16(false, a, false, bu, (short)0, cu, false, false);
  }
  int rbase = (lane >> 4) << 3;                    // D-layout: lanes 0-15 -> M=j, 16-31 -> M=8+j
#pragma unroll
  for (int j = 0; j < 8; ++j) {
    float g = cg[j], u = cu[j];
    float sg = g / (1.0f + __expf(-g));            // SiLU
    gated[(size_t)(slot0 + rbase + j) * FINT + f] = f2bf(u * sg);
  }
}

// ================= 5. grouped GEMM2 (down proj) ===============================
// grid.x = tile, grid.y = d-chunk of 128 (16 chunks covering DHID)
__global__ __launch_bounds__(256, 1)
void moe_down_kernel(const __bf16* __restrict__ gated, const __bf16* __restrict__ downT,
                     const int* __restrict__ tile_expert, const int* __restrict__ slot_tok,
                     const int* __restrict__ slot_k, const float* __restrict__ slot_w,
                     float* __restrict__ out4) {
  __shared__ __bf16 As[16 * FINT];                 // 24 KB
  int tile = blockIdx.x;
  int e = tile_expert[tile];
  if (e < 0) return;
  int tid = threadIdx.x, slot0 = tile * 16;
  for (int i = tid; i < 16 * 96; i += 256) {       // stage A tile (16 x 768 bf16)
    int r = i / 96, c8 = (i % 96) << 3;
    *(v8bf*)(As + r * FINT + c8) = *(const v8bf*)(gated + (size_t)(slot0 + r) * FINT + c8);
  }
  __syncthreads();
  int wave = tid >> 5, lane = tid & 31;
  int d = blockIdx.y * 128 + wave * 16 + (lane & 15);
  const __bf16* col  = downT + ((size_t)e * DHID + d) * FINT;
  const __bf16* arow = As + (size_t)(lane & 15) * FINT;
  v8f c = {};
  for (int k0 = 0; k0 < FINT; k0 += 32) {
    __builtin_prefetch(col + k0 + 256, 0, 0);
    v16bf a = load_a_frag(arow, k0, lane);
    v16bf b = load_b_frag(col, k0, lane);
    c = __builtin_amdgcn_wmma_f32_16x16x32_bf16(false, a, false, b, (short)0, c, false, false);
  }
  int rbase = (lane >> 4) << 3;
#pragma unroll
  for (int j = 0; j < 8; ++j) {
    int slot = slot0 + rbase + j;
    int k = slot_k[slot];
    if (k >= 0) {                                  // skip padding rows
      int tok = slot_tok[slot];
      float w = slot_w[slot];
      out4[((size_t)tok * TOPK + k) * DHID + d] = c[j] * w;   // unique (t,k) slot
    }
  }
}

// ================= 6. deterministic 4-way reduction per token =================
__global__ void moe_final_kernel(const float* __restrict__ out4, float* __restrict__ out) {
  size_t o = (size_t)blockIdx.x * 256 + threadIdx.x;   // T*D elements
  size_t t = o >> 11, d = o & (DHID - 1);
  const float* p = out4 + (t * TOPK) * DHID + d;
  float acc = p[0];
  acc += p[DHID];
  acc += p[2 * DHID];
  acc += p[3 * DHID];
  out[o] = acc;
}

// =================================== launch ===================================
extern "C" void kernel_launch(void* const* d_in, const int* in_sizes, int n_in,
                              void* d_out, int out_size, void* d_ws, size_t ws_size,
                              hipStream_t stream) {
  const float* x    = (const float*)d_in[0];   // (4,512,2048)
  const float* gate = (const float*)d_in[1];   // (2048,32)
  const float* gup  = (const float*)d_in[2];   // (32,2048,1536)
  const float* down = (const float*)d_in[3];   // (32,768,2048)
  float* out = (float*)d_out;

  // workspace carve-up (256B aligned)
  char* ws = (char*)d_ws;
  size_t off = 0;
  auto alloc = [&](size_t bytes) { void* p = ws + off; off = (off + bytes + 255) & ~(size_t)255; return p; };
  __bf16* xbf   = (__bf16*)alloc((size_t)T_TOK * DHID * 2);             // 8 MB
  __bf16* gupT  = (__bf16*)alloc((size_t)NEXP * TWO_F * DHID * 2);      // 192 MB
  __bf16* downT = (__bf16*)alloc((size_t)NEXP * DHID * FINT * 2);       // 96 MB
  __bf16* gated = (__bf16*)alloc((size_t)MAXSLOTS * FINT * 2);          // ~12.8 MB
  float*  out4  = (float*)alloc((size_t)T_TOK * TOPK * DHID * 4);       // 64 MB
  int*   topk_idx    = (int*)alloc((size_t)T_TOK * TOPK * 4);
  float* topk_w      = (float*)alloc((size_t)T_TOK * TOPK * 4);
  int*   counts      = (int*)alloc(NEXP * 4);
  int*   base        = (int*)alloc((NEXP + 1) * 4);
  int*   tile_expert = (int*)alloc(MAXTILES * 4);
  int*   slot_tok    = (int*)alloc(MAXSLOTS * 4);
  int*   slot_k      = (int*)alloc(MAXSLOTS * 4);
  float* slot_w      = (float*)alloc(MAXSLOTS * 4);
  (void)ws_size; (void)in_sizes; (void)n_in; (void)out_size;

  // 1. router (softmax + top-4 per token, one wave32 per token)
  router_kernel<<<T_TOK / 8, 256, 0, stream>>>(x, gate, topk_idx, topk_w);
  // 2. expert grouping (stable, padded to 16-row tiles)
  count_kernel<<<NEXP, 256, 0, stream>>>(topk_idx, counts);
  scan_kernel<<<1, 32, 0, stream>>>(counts, base, tile_expert);
  fill_kernel<<<NEXP, 256, 0, stream>>>(topk_idx, topk_w, base, slot_tok, slot_k, slot_w);
  // 3. bf16 conversion (weights transposed so WMMA B columns are K-contiguous)
  convert_x_kernel<<<(T_TOK * DHID) / 256, 256, 0, stream>>>(x, xbf);
  convert_gup_kernel<<<NEXP * TWO_F, 256, 0, stream>>>(gup, gupT);
  convert_down_kernel<<<NEXP * DHID, 256, 0, stream>>>(down, downT);
  // 4. gate_up GEMM + SiLU    (grid.y: 768 cols / (8 waves * 16) = 6)
  moe_gateup_kernel<<<dim3(MAXTILES, FINT / 128), 256, 0, stream>>>(
      xbf, gupT, tile_expert, slot_tok, gated);
  // 5. down GEMM, scaled, scattered to unique (token,k) rows (grid.y: 2048/128 = 16)
  moe_down_kernel<<<dim3(MAXTILES, DHID / 128), 256, 0, stream>>>(
      gated, downT, tile_expert, slot_tok, slot_k, slot_w, out4);
  // 6. fixed-order 4-way sum per token
  moe_final_kernel<<<(T_TOK * DHID) / 256, 256, 0, stream>>>(out4, out);
}
